// MultimodalTwoStreamsCrossAttentionModel_70643622084712
// MI455X (gfx1250) — matrix operand order repaired and checked by reading, not compiled, not run on hardware
//
#include <hip/hip_runtime.h>
#include <hip/hip_bf16.h>

// ---------------------------------------------------------------------------
// Types for CDNA5 WMMA (gfx1250, wave32)
// ---------------------------------------------------------------------------
typedef __bf16 bf16;
typedef __attribute__((ext_vector_type(16))) __bf16 v16bf;
typedef __attribute__((ext_vector_type(8)))  __bf16 v8bf;
typedef __attribute__((ext_vector_type(8)))  float  v8f;

#define DEVINL __device__ __forceinline__

DEVINL v16bf cat88(v8bf lo, v8bf hi) {
  return __builtin_shufflevector(lo, hi, 0,1,2,3,4,5,6,7,8,9,10,11,12,13,14,15);
}
DEVINL v8bf zero8() { v8bf z = {}; return z; }

DEVINL v8f wmma_bf16(v16bf a, v16bf b, v8f c) {
  // D(16x16,f32) = A(16x32,bf16) x B(32x16,bf16) + C
  return __builtin_amdgcn_wmma_f32_16x16x32_bf16(false, a, false, b, (short)0, c,
                                                 false, false);
}

// CDNA5 async global->LDS copy (ASYNCcnt path, bypasses VGPR round trip).
// lds_off = addrspace(3) byte offset (low 32 bits of generic __shared__ ptr).
DEVINL void async_ld128(unsigned lds_off, const bf16* g) {
  asm volatile("global_load_async_to_lds_b128 %0, %1, off"
               :: "v"(lds_off), "v"(g) : "memory");
}
DEVINL void async_wait0() { asm volatile("s_wait_asynccnt 0" ::: "memory"); }
DEVINL unsigned lds_addr(const void* p) { return (unsigned)(uintptr_t)p; }

// Problem constants
#define BB   2
#define TT   2048      // padded time length (power of two)
#define TPV  2047      // valid time steps after FUT truncation
#define MR   4096      // BB * TT rows
#define HIDC 128

// ---------------------------------------------------------------------------
// Pack (B, C, T) fp32 * mask  ->  (B*T, C) bf16 row-major  (32x32 LDS transpose)
// ---------------------------------------------------------------------------
__global__ __launch_bounds__(256)
void pack_transpose_mask(const float* __restrict__ src, const float* __restrict__ maskp,
                         int C, bf16* __restrict__ dst) {
  __shared__ float tile[32][33];
  int b  = blockIdx.z;
  int t0 = blockIdx.x * 32;
  int c0 = blockIdx.y * 32;
  int tx = threadIdx.x;          // 0..31
  int ty = threadIdx.y;          // 0..7
#pragma unroll
  for (int i = 0; i < 4; ++i) {
    int cl = ty + 8 * i;
    tile[cl][tx] = src[(size_t)(b * C + c0 + cl) * TT + t0 + tx];
  }
  __syncthreads();
#pragma unroll
  for (int i = 0; i < 4; ++i) {
    int tl = ty + 8 * i;
    float m = maskp[b * TT + t0 + tl];
    dst[(size_t)(b * TT + t0 + tl) * C + c0 + tx] = (bf16)(tile[tx][tl] * m);
  }
}

// ---------------------------------------------------------------------------
// Weight packers: fp32 (K,N) -> bf16 (N,K)   ;  conv fp32 (O,I,3) -> bf16 (O, 3*128)
// ---------------------------------------------------------------------------
__global__ void pack_wt(const float* __restrict__ w, int K, int N, bf16* __restrict__ wt) {
  int idx = blockIdx.x * 256 + threadIdx.x;
  if (idx >= N * K) return;
  int n = idx / K, k = idx - n * K;
  wt[idx] = (bf16)w[(size_t)k * N + n];
}

__global__ void pack_conv_wt(const float* __restrict__ w, bf16* __restrict__ wt) {
  int idx = blockIdx.x * 256 + threadIdx.x;   // over 128*384
  if (idx >= 128 * 384) return;
  int o = idx / 384, kk = idx - o * 384;
  int tap = kk >> 7, i = kk & 127;
  wt[idx] = (bf16)w[(size_t)(o * 128 + i) * 3 + tap];
}

// ---------------------------------------------------------------------------
// WMMA GEMM: Y = act(A(MxK,bf16) @ Wt^T + bias), Wt stored (N,K) bf16.
// 64x64 tile / block (4 waves, each wave 2x2 WMMA 16x16 tiles), K chunk 32.
// Tiles staged with global_load_async_to_lds_b128 (ASYNCcnt).
// ---------------------------------------------------------------------------
__global__ __launch_bounds__(128)
void gemm_bias_act(const bf16* __restrict__ A, int lda,
                   const bf16* __restrict__ Wt,
                   const float* __restrict__ bias,
                   int M, int N, int K, int relu,
                   bf16* __restrict__ outB, int ldoB, int coloff,
                   float* __restrict__ outF, int ldoF) {
  __shared__ bf16 As[64][32];
  __shared__ bf16 Bs[64][32];
  const int tid  = threadIdx.x;
  const int lane = tid & 31;
  const int wv   = tid >> 5;
  const int wr   = wv >> 1, wc = wv & 1;
  const int m0   = blockIdx.y * 64;
  const int n0   = blockIdx.x * 64;
  const int lrow = tid >> 1;
  const int lcol = (tid & 1) * 16;
  const int lm    = lane & 15;
  const int hi    = lane >> 4;
  const int kb    = hi ? 8 : 0;
  const int kbase = hi ? 16 : 0;
  const unsigned ldsA = lds_addr(&As[lrow][lcol]);
  const unsigned ldsB = lds_addr(&Bs[lrow][lcol]);

  v8f acc[2][2];
#pragma unroll
  for (int a = 0; a < 2; ++a)
#pragma unroll
    for (int b = 0; b < 2; ++b) { v8f z = {}; acc[a][b] = z; }

  for (int k0 = 0; k0 < K; k0 += 32) {
    // stage A tile (async direct-to-LDS)
    {
      int gr = m0 + lrow;
      if (gr < M) {
        const bf16* p = A + (size_t)gr * lda + k0 + lcol;
        async_ld128(ldsA, p);
        async_ld128(ldsA + 16u, p + 8);
        if (k0 + 32 < K) __builtin_prefetch(p + 32, 0, 1);
      } else {
        *(v8bf*)&As[lrow][lcol]     = zero8();
        *(v8bf*)&As[lrow][lcol + 8] = zero8();
      }
    }
    // stage B tile (weights, (N,K) layout)
    {
      int gn = n0 + lrow;
      if (gn < N) {
        const bf16* p = Wt + (size_t)gn * K + k0 + lcol;
        async_ld128(ldsB, p);
        async_ld128(ldsB + 16u, p + 8);
      } else {
        *(v8bf*)&Bs[lrow][lcol]     = zero8();
        *(v8bf*)&Bs[lrow][lcol + 8] = zero8();
      }
    }
    async_wait0();
    __syncthreads();

    v16bf af[2], bfr[2];
#pragma unroll
    for (int tm = 0; tm < 2; ++tm) {
      const bf16* r = &As[wr * 32 + tm * 16 + lm][0];
      af[tm] = cat88(*(const v8bf*)(r + kbase), *(const v8bf*)(r + kbase + 8));
    }
#pragma unroll
    for (int tn = 0; tn < 2; ++tn) {
      const bf16* r = &Bs[wc * 32 + tn * 16 + lm][0];
      bfr[tn] = cat88(*(const v8bf*)(r + kbase), *(const v8bf*)(r + kbase + 8));
    }
#pragma unroll
    for (int tm = 0; tm < 2; ++tm)
#pragma unroll
      for (int tn = 0; tn < 2; ++tn)
        acc[tm][tn] = wmma_bf16(af[tm], bfr[tn], acc[tm][tn]);
    __syncthreads();
  }

  // epilogue: bias + activation + store (D layout: row = r + 8*hi, col = lm)
#pragma unroll
  for (int tn = 0; tn < 2; ++tn) {
    int gn = n0 + wc * 32 + tn * 16 + lm;
    float bv = (bias && gn < N) ? bias[gn] : 0.f;
#pragma unroll
    for (int tm = 0; tm < 2; ++tm) {
#pragma unroll
      for (int r = 0; r < 8; ++r) {
        int gm = m0 + wr * 32 + tm * 16 + r + hi * 8;
        if (gm < M && gn < N) {
          float v = acc[tm][tn][r] + bv;
          if (relu) v = fmaxf(v, 0.f);
          if (outB) outB[(size_t)gm * ldoB + coloff + gn] = (bf16)v;
          if (outF) outF[(size_t)gm * ldoF + gn] = v;
        }
      }
    }
  }
}

// ---------------------------------------------------------------------------
// Causal dilated conv as implicit-im2col WMMA GEMM.
// X: (B*TT, 128) bf16, Wc: (128, 384) bf16 with K index = tap*128 + in_ch.
// y[t] = W0 x[t-2d] + W1 x[t-d] + W2 x[t] + b  (zero below t=0, per batch)
// ---------------------------------------------------------------------------
__global__ __launch_bounds__(128)
void conv_gemm(const bf16* __restrict__ X, const bf16* __restrict__ Wc,
               const float* __restrict__ bias, int dil, bf16* __restrict__ Y) {
  __shared__ bf16 As[64][32];
  __shared__ bf16 Bs[64][32];
  const int tid  = threadIdx.x;
  const int lane = tid & 31;
  const int wv   = tid >> 5;
  const int wr   = wv >> 1, wc = wv & 1;
  const int m0   = blockIdx.y * 64;
  const int n0   = blockIdx.x * 64;
  const int lrow = tid >> 1;
  const int lcol = (tid & 1) * 16;
  const int lm    = lane & 15;
  const int hi    = lane >> 4;
  const int kb    = hi ? 8 : 0;
  const int kbase = hi ? 16 : 0;
  const unsigned ldsA = lds_addr(&As[lrow][lcol]);
  const unsigned ldsB = lds_addr(&Bs[lrow][lcol]);

  v8f acc[2][2];
#pragma unroll
  for (int a = 0; a < 2; ++a)
#pragma unroll
    for (int b = 0; b < 2; ++b) { v8f z = {}; acc[a][b] = z; }

  for (int k0 = 0; k0 < 384; k0 += 32) {
    int tap   = k0 >> 7;
    int shift = (2 - tap) * dil;
    int ch0   = k0 & 127;
    {
      int gr = m0 + lrow;
      int b  = gr >> 11, t = gr & (TT - 1);
      int ts = t - shift;
      if (ts >= 0) {
        const bf16* p = X + (size_t)((b << 11) + ts) * HIDC + ch0 + lcol;
        async_ld128(ldsA, p);
        async_ld128(ldsA + 16u, p + 8);
      } else {
        *(v8bf*)&As[lrow][lcol]     = zero8();
        *(v8bf*)&As[lrow][lcol + 8] = zero8();
      }
    }
    {
      int gn = n0 + lrow;
      const bf16* p = Wc + (size_t)gn * 384 + k0 + lcol;
      async_ld128(ldsB, p);
      async_ld128(ldsB + 16u, p + 8);
    }
    async_wait0();
    __syncthreads();

    v16bf af[2], bfr[2];
#pragma unroll
    for (int tm = 0; tm < 2; ++tm) {
      const bf16* r = &As[wr * 32 + tm * 16 + lm][0];
      af[tm] = cat88(*(const v8bf*)(r + kbase), *(const v8bf*)(r + kbase + 8));
    }
#pragma unroll
    for (int tn = 0; tn < 2; ++tn) {
      const bf16* r = &Bs[wc * 32 + tn * 16 + lm][0];
      bfr[tn] = cat88(*(const v8bf*)(r + kbase), *(const v8bf*)(r + kbase + 8));
    }
#pragma unroll
    for (int tm = 0; tm < 2; ++tm)
#pragma unroll
      for (int tn = 0; tn < 2; ++tn)
        acc[tm][tn] = wmma_bf16(af[tm], bfr[tn], acc[tm][tn]);
    __syncthreads();
  }

#pragma unroll
  for (int tn = 0; tn < 2; ++tn) {
    int gn = n0 + wc * 32 + tn * 16 + lm;
    float bv = bias[gn];
#pragma unroll
    for (int tm = 0; tm < 2; ++tm)
#pragma unroll
      for (int r = 0; r < 8; ++r) {
        int gm = m0 + wr * 32 + tm * 16 + r + hi * 8;
        Y[(size_t)gm * HIDC + gn] = (bf16)(acc[tm][tn][r] + bv);
      }
  }
}

// ---------------------------------------------------------------------------
// Depthwise 3-tap conv (same pad, dil=1) + channel LayerNorm. One block per row.
// toff shifts the source time index (FUT shift for the text stream).
// ---------------------------------------------------------------------------
__global__ __launch_bounds__(128)
void dwln_kernel(const bf16* __restrict__ X, const float* __restrict__ dw,
                 const float* __restrict__ g, const float* __restrict__ bta,
                 int toff, bf16* __restrict__ Y) {
  int row = blockIdx.x;
  int c   = threadIdx.x;
  int b = row >> 11, t = row & (TT - 1);
  float acc = 0.f;
  if (t < TPV) {
#pragma unroll
    for (int k = 0; k < 3; ++k) {
      int ts = t + k - 1;
      if (ts >= 0 && ts < TPV)
        acc += dw[c * 3 + k] * (float)X[(size_t)((b << 11) + ts + toff) * HIDC + c];
    }
  }
  float s1 = acc, s2 = acc * acc;
#pragma unroll
  for (int m = 16; m >= 1; m >>= 1) {
    s1 += __shfl_xor(s1, m, 32);
    s2 += __shfl_xor(s2, m, 32);
  }
  __shared__ float ws1[4], ws2[4];
  int w = threadIdx.x >> 5;
  if ((threadIdx.x & 31) == 0) { ws1[w] = s1; ws2[w] = s2; }
  __syncthreads();
  float tot  = ws1[0] + ws1[1] + ws1[2] + ws1[3];
  float tot2 = ws2[0] + ws2[1] + ws2[2] + ws2[3];
  float mu  = tot * (1.f / 128.f);
  float var = tot2 * (1.f / 128.f) - mu * mu;
  float y = (acc - mu) * rsqrtf(var + 1e-5f) * g[c] + bta[c];
  Y[(size_t)row * HIDC + c] = (t < TPV) ? (bf16)y : (bf16)0.f;
}

// ---------------------------------------------------------------------------
// Local causal windowed attention (HEADS=4, dh=32) via WMMA.
// One wave per (batch, head, 16-query block). Q/K/V bf16 (B*TT, 128).
// ---------------------------------------------------------------------------
template <int WIN>
__global__ __launch_bounds__(128)
void local_attn(const bf16* __restrict__ Q, const bf16* __restrict__ Kk,
                const bf16* __restrict__ Vv, bf16* __restrict__ O) {
  constexpr int NBLK  = WIN / 16 + 2;   // even; covers keys [q0-WIN, q0+31]
  constexpr int NPAIR = NBLK / 2;
  __shared__ bf16 P[4][16][16 * NBLK];

  const int ws   = threadIdx.x >> 5;
  const int lane = threadIdx.x & 31;
  const int wid  = blockIdx.x * 4 + ws;
  const int qb = wid & 127;
  const int h  = (wid >> 7) & 3;
  const int b  = wid >> 9;
  const int q0 = qb << 4;
  const int lm    = lane & 15;
  const int hi    = lane >> 4;
  const int kb    = hi ? 8 : 0;
  const int kbase = hi ? 16 : 0;

  // Q fragment (A operand): row q0+lm, channels {kbase..kbase+15}
  const bf16* qrow = Q + (size_t)((b << 11) + q0 + lm) * HIDC + h * 32;
  v16bf qf = cat88(*(const v8bf*)(qrow + kbase), *(const v8bf*)(qrow + kbase + 8));

  // S = Q K^T over NBLK key blocks
  v8f S[NBLK];
#pragma unroll
  for (int j = 0; j < NBLK; ++j) {
    int key = q0 - WIN + j * 16 + lm;
    v16bf kf;
    if (key >= 0 && key < TPV) {
      const bf16* kr = Kk + (size_t)((b << 11) + key) * HIDC + h * 32 + kbase;
      kf = cat88(*(const v8bf*)kr, *(const v8bf*)(kr + 8));
    } else {
      kf = cat88(zero8(), zero8());
    }
    v8f z = {};
    S[j] = wmma_bf16(qf, kf, z);
  }

  // mask + softmax (per query row = r + 8*hi; reduce across 16-lane group)
  const float scale = 0.17677669529663689f;  // 1/sqrt(32)
  float rmax[8];
#pragma unroll
  for (int r = 0; r < 8; ++r) rmax[r] = -3.0e38f;
#pragma unroll
  for (int j = 0; j < NBLK; ++j) {
    int key = q0 - WIN + j * 16 + lm;
#pragma unroll
    for (int r = 0; r < 8; ++r) {
      int q = q0 + r + hi * 8;
      float sv = S[j][r] * scale;
      bool valid = (key >= 0) && (key < TPV) && (key > q - WIN) && (key <= q);
      sv = valid ? sv : -1.0e30f;
      S[j][r] = sv;
      rmax[r] = fmaxf(rmax[r], sv);
    }
  }
#pragma unroll
  for (int r = 0; r < 8; ++r)
#pragma unroll
    for (int m = 8; m >= 1; m >>= 1) rmax[r] = fmaxf(rmax[r], __shfl_xor(rmax[r], m, 32));
  float rsum[8];
#pragma unroll
  for (int r = 0; r < 8; ++r) rsum[r] = 0.f;
#pragma unroll
  for (int j = 0; j < NBLK; ++j)
#pragma unroll
    for (int r = 0; r < 8; ++r) {
      float p = __expf(S[j][r] - rmax[r]);
      S[j][r] = p;
      rsum[r] += p;
    }
  float rinv[8];
#pragma unroll
  for (int r = 0; r < 8; ++r) {
#pragma unroll
    for (int m = 8; m >= 1; m >>= 1) rsum[r] += __shfl_xor(rsum[r], m, 32);
    rinv[r] = 1.f / fmaxf(rsum[r], 1e-20f);
  }

  // stage probs to LDS, re-layout into A operand for P@V
#pragma unroll
  for (int j = 0; j < NBLK; ++j)
#pragma unroll
    for (int r = 0; r < 8; ++r)
      P[ws][r + hi * 8][j * 16 + lm] = (bf16)S[j][r];
  __syncthreads();

  v8f oa[2];
  { v8f z = {}; oa[0] = z; oa[1] = z; }
#pragma unroll
  for (int jj = 0; jj < NPAIR; ++jj) {
    const bf16* pr = &P[ws][lm][jj * 32];
    v16bf pf = cat88(*(const v8bf*)(pr + kbase), *(const v8bf*)(pr + kbase + 8));
    int kg0 = q0 - WIN + jj * 32 + kbase;
#pragma unroll
    for (int hh = 0; hh < 2; ++hh) {
      v16bf vf;
#pragma unroll
      for (int e = 0; e < 16; ++e) {
        int key = kg0 + e;
        float val = (key >= 0 && key < TPV)
                        ? (float)Vv[(size_t)((b << 11) + key) * HIDC + h * 32 + hh * 16 + lm]
                        : 0.f;
        vf[e] = (bf16)val;
      }
      oa[hh] = wmma_bf16(pf, vf, oa[hh]);
    }
  }

#pragma unroll
  for (int hh = 0; hh < 2; ++hh)
#pragma unroll
    for (int r = 0; r < 8; ++r) {
      int q = q0 + r + hi * 8;
      O[(size_t)((b << 11) + q) * HIDC + h * 32 + hh * 16 + lm] = (bf16)(oa[hh][r] * rinv[r]);
    }
}

// ---------------------------------------------------------------------------
// Final: (B*TT, 64) fp32 -> (B, 64, TT) with FUT pad at t=0 and mask multiply
// ---------------------------------------------------------------------------
__global__ void final_out_kernel(const float* __restrict__ y, const float* __restrict__ maskp,
                                 float* __restrict__ out) {
  int idx = blockIdx.x * 256 + threadIdx.x;
  if (idx >= BB * 64 * TT) return;
  int t = idx & (TT - 1);
  int b = idx >> 17;
  int o = (idx >> 11) & 63;
  float v = 0.f;
  if (t >= 1) v = y[(size_t)((b << 11) + (t - 1)) * 64 + o] * maskp[(b << 11) + t];
  out[idx] = v;
}

// ---------------------------------------------------------------------------
// Host orchestration
// ---------------------------------------------------------------------------
extern "C" void kernel_launch(void* const* d_in, const int* in_sizes, int n_in,
                              void* d_out, int out_size, void* d_ws, size_t ws_size,
                              hipStream_t stream) {
  (void)in_sizes; (void)n_in; (void)out_size; (void)ws_size;
  const float* visual = (const float*)d_in[0];
  const float* text   = (const float*)d_in[1];
  const float* maskp  = (const float*)d_in[2];
  // Flattened params, jax pytree order (dict keys sorted):
  // 3..19 attn_l[0]: bk bo bq bv k_b k_dw k_g q_b q_dw q_g v_b v_dw v_g wk wo wq wv
  // 20..36 attn_s[0] (same order), 37..42 dconvs b0 w0 b1 w1 b2 w2,
  // 43,44 fc1 b/w, 45,46 fc2 b/w, 47,48 fc_l b/w, 49,50 fc_s b/w,
  // 51..56 txt_enc b0 w0 b1 w1 b2 w2, 57..62 vis_enc b0 w0 b1 w1 b2 w2
  const int AL = 3, AS = 20, DC = 37, FC1 = 43, FC2 = 45, FCL = 47, FCS = 49,
            TXE = 51, VIE = 57;
  auto F = [&](int i) { return (const float*)d_in[i]; };

  char* base = (char*)d_ws;
  size_t off = 0;
  auto ALLOC = [&](size_t bytes) -> char* {
    char* p = base + off;
    off = (off + bytes + 255) & ~(size_t)255;
    return p;
  };

  bf16* Xbig   = (bf16*)ALLOC((size_t)MR * 1024 * 2);  // vis pack, then txt pack
  bf16* Tmid   = (bf16*)ALLOC((size_t)MR * 768 * 2);
  bf16* s0 = (bf16*)ALLOC((size_t)MR * HIDC * 2);
  bf16* s1 = (bf16*)ALLOC((size_t)MR * HIDC * 2);
  bf16* s2 = (bf16*)ALLOC((size_t)MR * HIDC * 2);
  bf16* s3 = (bf16*)ALLOC((size_t)MR * HIDC * 2);
  bf16* s4 = (bf16*)ALLOC((size_t)MR * HIDC * 2);
  bf16* s5 = (bf16*)ALLOC((size_t)MR * HIDC * 2);
  bf16* vstream = (bf16*)ALLOC((size_t)MR * HIDC * 2);
  bf16* tfeat   = (bf16*)ALLOC((size_t)MR * HIDC * 2);
  bf16* catb    = (bf16*)ALLOC((size_t)MR * 256 * 2);
  float* f2     = (float*)ALLOC((size_t)MR * 64 * 4);
  // packed weights
  bf16* wv1t = (bf16*)ALLOC(128 * 1024 * 2);
  bf16* wv2t = (bf16*)ALLOC(128 * 128 * 2);
  bf16* wv3t = (bf16*)ALLOC(128 * 128 * 2);
  bf16* wt1t = (bf16*)ALLOC(768 * 768 * 2);
  bf16* wt2t = (bf16*)ALLOC(128 * 768 * 2);
  bf16* wt3t = (bf16*)ALLOC(128 * 128 * 2);
  bf16* wc1t = (bf16*)ALLOC(128 * 384 * 2);
  bf16* wc2t = (bf16*)ALLOC(128 * 384 * 2);
  bf16* wc3t = (bf16*)ALLOC(128 * 384 * 2);
  bf16* fc1t = (bf16*)ALLOC(128 * 256 * 2);
  bf16* fc2t = (bf16*)ALLOC(64 * 128 * 2);
  bf16* wqt = (bf16*)ALLOC(128 * 128 * 2);  // per-stream, reused
  bf16* wkt = (bf16*)ALLOC(128 * 128 * 2);
  bf16* wvt = (bf16*)ALLOC(128 * 128 * 2);
  bf16* wot = (bf16*)ALLOC(128 * 128 * 2);
  bf16* fct = (bf16*)ALLOC(128 * 128 * 2);

  auto packw = [&](const float* w, int K, int N, bf16* dst) {
    pack_wt<<<(N * K + 255) / 256, 256, 0, stream>>>(w, K, N, dst);
  };
  auto gemm = [&](const bf16* A, int lda, const bf16* Wt, const float* bias, int N,
                  int K, int relu, bf16* outB, int ldoB, int coloff, float* outF,
                  int ldoF) {
    dim3 g((N + 63) / 64, MR / 64);
    gemm_bias_act<<<g, 128, 0, stream>>>(A, lda, Wt, bias, MR, N, K, relu, outB,
                                         ldoB, coloff, outF, ldoF);
  };

  // ---- pack static weights
  packw(F(VIE + 1), 1024, 128, wv1t);
  packw(F(VIE + 3), 128, 128, wv2t);
  packw(F(VIE + 5), 128, 128, wv3t);
  packw(F(TXE + 1), 768, 768, wt1t);
  packw(F(TXE + 3), 768, 128, wt2t);
  packw(F(TXE + 5), 128, 128, wt3t);
  packw(F(FC1 + 1), 256, 128, fc1t);
  packw(F(FC2 + 1), 128, 64, fc2t);
  pack_conv_wt<<<(128 * 384 + 255) / 256, 256, 0, stream>>>(F(DC + 1), wc1t);
  pack_conv_wt<<<(128 * 384 + 255) / 256, 256, 0, stream>>>(F(DC + 3), wc2t);
  pack_conv_wt<<<(128 * 384 + 255) / 256, 256, 0, stream>>>(F(DC + 5), wc3t);

  // ---- visual branch: pack + MLP + 3 causal convs
  {
    dim3 g(TT / 32, 1024 / 32, BB);
    pack_transpose_mask<<<g, dim3(32, 8), 0, stream>>>(visual, maskp, 1024, Xbig);
  }
  gemm(Xbig, 1024, wv1t, F(VIE + 0), 128, 1024, 1, s0, 128, 0, nullptr, 0);
  gemm(s0, 128, wv2t, F(VIE + 2), 128, 128, 1, s1, 128, 0, nullptr, 0);
  gemm(s1, 128, wv3t, F(VIE + 4), 128, 128, 1, s0, 128, 0, nullptr, 0);
  conv_gemm<<<dim3(2, MR / 64), 128, 0, stream>>>(s0, wc1t, F(DC + 0), 1, s1);
  conv_gemm<<<dim3(2, MR / 64), 128, 0, stream>>>(s1, wc2t, F(DC + 2), 2, s0);
  conv_gemm<<<dim3(2, MR / 64), 128, 0, stream>>>(s0, wc3t, F(DC + 4), 4, vstream);

  // ---- text branch
  {
    dim3 g(TT / 32, 768 / 32, BB);
    pack_transpose_mask<<<g, dim3(32, 8), 0, stream>>>(text, maskp, 768, Xbig);
  }
  gemm(Xbig, 768, wt1t, F(TXE + 0), 768, 768, 1, Tmid, 768, 0, nullptr, 0);
  gemm(Tmid, 768, wt2t, F(TXE + 2), 128, 768, 1, s0, 128, 0, nullptr, 0);
  gemm(s0, 128, wt3t, F(TXE + 4), 128, 128, 1, tfeat, 128, 0, nullptr, 0);

  // ---- two attention streams
  auto run_stream = [&](int P, int FCB, int win, int coloff) {
    packw(F(P + 15), 128, 128, wqt);
    packw(F(P + 13), 128, 128, wkt);
    packw(F(P + 16), 128, 128, wvt);
    packw(F(P + 14), 128, 128, wot);
    packw(F(FCB + 1), 128, 128, fct);
    dwln_kernel<<<MR, 128, 0, stream>>>(tfeat,   F(P + 8), F(P + 9), F(P + 7), 1, s0);
    dwln_kernel<<<MR, 128, 0, stream>>>(vstream, F(P + 5), F(P + 6), F(P + 4), 0, s1);
    dwln_kernel<<<MR, 128, 0, stream>>>(vstream, F(P + 11), F(P + 12), F(P + 10), 0, s2);
    gemm(s0, 128, wqt, F(P + 2), 128, 128, 0, s3, 128, 0, nullptr, 0);
    gemm(s1, 128, wkt, F(P + 0), 128, 128, 0, s4, 128, 0, nullptr, 0);
    gemm(s2, 128, wvt, F(P + 3), 128, 128, 0, s5, 128, 0, nullptr, 0);
    if (win == 32)
      local_attn<32><<<256, 128, 0, stream>>>(s3, s4, s5, s0);
    else
      local_attn<128><<<256, 128, 0, stream>>>(s3, s4, s5, s0);
    gemm(s0, 128, wot, F(P + 1), 128, 128, 0, s1, 128, 0, nullptr, 0);
    gemm(s1, 128, fct, F(FCB + 0), 128, 128, 1, catb, 256, coloff, nullptr, 0);
  };
  run_stream(AS, FCS, 32, 0);     // short-window stream -> cat[:, 0:128]
  run_stream(AL, FCL, 128, 128);  // long-window stream  -> cat[:, 128:256]

  // ---- head: fc1(256->128, relu), fc2(128->64), transpose+pad+mask
  gemm(catb, 256, fc1t, F(FC1 + 0), 128, 256, 1, s0, 128, 0, nullptr, 0);
  gemm(s0, 128, fc2t, F(FC2 + 0), 64, 128, 0, nullptr, 0, 0, f2, 64);
  final_out_kernel<<<(BB * 64 * TT + 255) / 256, 256, 0, stream>>>(f2, maskp,
                                                                   (float*)d_out);
}
